// ACTRNNCell_79044578116290
// MI455X (gfx1250) — matrix operand order repaired and checked by reading, not compiled, not run on hardware
//
#include <hip/hip_runtime.h>

#define BDIM 4096
#define HDIM 1024
#define GDIM 3072          // 3*H
#define DEPTH 10
#define KCHUNK 256
#define LROW 264           // padded LDS row pitch (bf16 elems): 528B -> bank stride 4 dwords, conflict-free

typedef __attribute__((ext_vector_type(16))) __bf16 v16bf;
typedef __attribute__((ext_vector_type(8)))  float  v8f;
typedef unsigned short ushort_t;

union Frag { v16bf v; uint4 q[2]; };

__device__ __forceinline__ v8f wmma_bf16(v16bf a, v16bf b, v8f c) {
  // (neg_a, A, neg_b, B, c_mod, C, reuse_a, reuse_b)
  return __builtin_amdgcn_wmma_f32_16x16x32_bf16(false, a, false, b, (short)0, c, false, false);
}

__device__ __forceinline__ ushort_t f2bf(float f) {
  unsigned u = __float_as_uint(f);
  u += 0x7fffu + ((u >> 16) & 1u);   // round-to-nearest-even
  return (ushort_t)(u >> 16);
}

__device__ __forceinline__ float sigm(float x) { return 1.f / (1.f + expf(-x)); }

// ---------------- small utility kernels ----------------
__global__ void k_zero(float* __restrict__ p, int n) {
  int i = blockIdx.x * 256 + threadIdx.x;
  if (i < n) p[i] = 0.f;
}
__global__ void k_cvt(const float* __restrict__ s, ushort_t* __restrict__ d, int n) {
  int i = blockIdx.x * 256 + threadIdx.x;
  if (i < n) d[i] = f2bf(s[i]);
}
// pack W_ih[:, 0:H] (row stride H+1) into dense bf16 (row stride H)
__global__ void k_cvt_wih(const float* __restrict__ w, ushort_t* __restrict__ d) {
  int i = blockIdx.x * 256 + threadIdx.x;     // over GDIM*HDIM
  int r = i >> 10, c = i & 1023;
  d[i] = f2bf(w[(size_t)r * (HDIM + 1) + c]);
}
__global__ void k_inith(const float* __restrict__ h0, float* __restrict__ hF,
                        ushort_t* __restrict__ hB, int n) {
  int i = blockIdx.x * 256 + threadIdx.x;
  if (i < n) { float v = h0[i]; hF[i] = v; hB[i] = f2bf(v); }
}
__global__ void k_copy(float* __restrict__ d, const float* __restrict__ s, int n) {
  int i = blockIdx.x * 256 + threadIdx.x;
  if (i < n) d[i] = s[i];
}

// ---------------- ACT halting + mean-field accumulation ----------------
// one wave32 per batch row; runs BEFORE the GRU update of the same depth
__global__ void k_halt(const float* __restrict__ H_, const float* __restrict__ wHalt,
                       const float* __restrict__ bHalt, float* __restrict__ accum,
                       float* __restrict__ merged, float* __restrict__ hpAcc,
                       float* __restrict__ numUpd, int d) {
  const int wave = threadIdx.x >> 5, lane = threadIdx.x & 31;
  const int b = blockIdx.x * 8 + wave;
  const float* hrow = H_ + (size_t)b * HDIM;
  float s = 0.f;
  for (int i = lane; i < HDIM; i += 32) s += hrow[i] * wHalt[i];
  for (int m = 16; m; m >>= 1) s += __shfl_xor(s, m, 32);   // all lanes hold full dot
  const float acc = accum[b];
  const float p = sigm(s + bHalt[0]);
  const float alivePre = (acc < 1.f) ? 1.f : 0.f;
  float nh = (p * alivePre + acc > 0.9f) ? 1.f : 0.f;
  if (d == DEPTH - 1) nh = 1.f;
  const float rem = 1.f - acc + 1e-15f;
  const float alive = alivePre * (1.f - nh);
  const float sp = p * alive + rem * nh;
  if (lane == 0) {
    accum[b] = acc + sp;
    hpAcc[b] += sp * alive;
    numUpd[b] += alive;
  }
  float* mrow = merged + (size_t)b * HDIM;
  for (int i = lane; i < HDIM; i += 32) mrow[i] += sp * hrow[i];   // emit pre-update hidden
}

// ---------------- fused bf16-WMMA GEMM ----------------
// Wave tile: 32 rows x 32 cols x 3 gates (12 v8f accums). Workgroup: 8 waves => 256 rows.
// mode 0: Gi_base = A @ Bw^T + b_ih           (A = inputs_bf16, Bw = W_ih_bf16)
// mode 1: gh = A @ Bw^T; fused GRU epilogue   (A = h_bf16,      Bw = W_hh_bf16)
__global__ void __launch_bounds__(256)
k_gemm(const ushort_t* __restrict__ A, const ushort_t* __restrict__ Bw,
       int mode, float flag,
       float* __restrict__ GiOut, const float* __restrict__ biasIh,
       const float* __restrict__ GiIn, const float* __restrict__ biasHh,
       const float* __restrict__ WihFull, const float* __restrict__ Hcur,
       float* __restrict__ Hnext, ushort_t* __restrict__ HnextBf) {
  __shared__ ushort_t ldsB[96 * LROW];   // 3 gates x 32 cols x KCHUNK, padded (50,688 B)

  const int tid = threadIdx.x;
  const int wave = tid >> 5;
  const int lane = tid & 31;
  const int laneHi = lane >> 4;          // 0/1 half of wave32
  const int laneLo = lane & 15;

  const int n0 = blockIdx.x * 32;                  // hidden-column group [0,1024)
  const int rowBase = blockIdx.y * 256 + wave * 32;

  v8f acc[3][2][2];                                // [gate][ntile][mtile]
  const v8f vz = {0.f, 0.f, 0.f, 0.f, 0.f, 0.f, 0.f, 0.f};
#pragma unroll
  for (int g = 0; g < 3; ++g)
#pragma unroll
    for (int nt = 0; nt < 2; ++nt)
#pragma unroll
      for (int mt = 0; mt < 2; ++mt) acc[g][nt][mt] = vz;

  for (int kc = 0; kc < HDIM; kc += KCHUNK) {
    // cooperative stage of W panel chunk: 96 rows x 256 bf16 -> 3072 uint4
    for (int i = tid; i < 96 * 32; i += 256) {
      const int p = i >> 5;                        // panel row (gate*32 + col)
      const int kk = (i & 31) << 3;                // bf16 offset within chunk
      const int g = p >> 5, c = p & 31;
      const ushort_t* src = Bw + (size_t)(g * HDIM + n0 + c) * HDIM + kc + kk;
      *(uint4*)&ldsB[p * LROW + kk] = *(const uint4*)src;
    }
    __syncthreads();

#pragma unroll
    for (int k = 0; k < KCHUNK; k += 32) {
      // A fragments (global, 2 M-tiles)
      Frag a[2];
#pragma unroll
      for (int mt = 0; mt < 2; ++mt) {
        // 16-bit A layout: lanes lo: K=k..k+7 & k+16..k+23 ; lanes hi: +8
        const ushort_t* ap =
            A + (size_t)(rowBase + mt * 16 + laneLo) * HDIM + kc + k + 8 * laneHi;
        a[mt].q[0] = *(const uint4*)ap;
        a[mt].q[1] = *(const uint4*)(ap + 16);
      }
      // B fragments (LDS) into *distinct* registers so all 12 ds_load_b128 can be
      // clause-grouped and waited once, leaving the 12 WMMAs to issue back-to-back.
      Frag b[3][2];
#pragma unroll
      for (int g = 0; g < 3; ++g)
#pragma unroll
        for (int nt = 0; nt < 2; ++nt) {
          // B layout: lane = N, lanes lo hold K=k..k+7 & k+16..k+23 ; lanes hi: +8
          const ushort_t* bp = &ldsB[(g * 32 + nt * 16 + laneLo) * LROW + k + 8 * laneHi];
          b[g][nt].q[0] = *(const uint4*)bp;
          b[g][nt].q[1] = *(const uint4*)(bp + 16);
        }
#pragma unroll
      for (int g = 0; g < 3; ++g)
#pragma unroll
        for (int nt = 0; nt < 2; ++nt) {
          acc[g][nt][0] = wmma_bf16(a[0].v, b[g][nt].v, acc[g][nt][0]);
          acc[g][nt][1] = wmma_bf16(a[1].v, b[g][nt].v, acc[g][nt][1]);
        }
    }
    __syncthreads();
  }

  // ---------------- epilogue ----------------
  // C layout: element r of v8f -> row M = r + 8*laneHi, col N = laneLo
  if (mode == 0) {
#pragma unroll
    for (int nt = 0; nt < 2; ++nt) {
      const int col = n0 + nt * 16 + laneLo;
#pragma unroll
      for (int g = 0; g < 3; ++g) {
        const float bi = biasIh[g * HDIM + col];
#pragma unroll
        for (int mt = 0; mt < 2; ++mt) {
          const int rb = rowBase + mt * 16 + 8 * laneHi;
#pragma unroll
          for (int r = 0; r < 8; ++r)
            GiOut[(size_t)(rb + r) * GDIM + g * HDIM + col] = acc[g][nt][mt][r] + bi;
        }
      }
    }
  } else {
#pragma unroll
    for (int nt = 0; nt < 2; ++nt) {
      const int col = n0 + nt * 16 + laneLo;
      // b_hh plus the per-depth flag column of W_ih (commutes into the pre-activation)
      const float bhr = biasHh[col]            + flag * WihFull[(size_t)(col)            * (HDIM + 1) + HDIM];
      const float bhz = biasHh[HDIM + col]     + flag * WihFull[(size_t)(HDIM + col)     * (HDIM + 1) + HDIM];
      const float bhn = biasHh[2 * HDIM + col] + flag * WihFull[(size_t)(2 * HDIM + col) * (HDIM + 1) + HDIM];
#pragma unroll
      for (int mt = 0; mt < 2; ++mt) {
        const int rb = rowBase + mt * 16 + 8 * laneHi;
#pragma unroll
        for (int r = 0; r < 8; ++r) {
          const int row = rb + r;
          const size_t gio = (size_t)row * GDIM;
          const float ghr = acc[0][nt][mt][r] + bhr;
          const float ghz = acc[1][nt][mt][r] + bhz;
          const float ghn = acc[2][nt][mt][r] + bhn;
          const float rg = sigm(GiIn[gio + col] + ghr);
          const float zg = sigm(GiIn[gio + HDIM + col] + ghz);
          const float ng = tanhf(GiIn[gio + 2 * HDIM + col] + rg * ghn);
          const float hp = Hcur[(size_t)row * HDIM + col];
          const float hn = (1.f - zg) * ng + zg * hp;
          Hnext[(size_t)row * HDIM + col] = hn;
          HnextBf[(size_t)row * HDIM + col] = f2bf(hn);
        }
      }
    }
  }
}

// ---------------- host orchestration ----------------
extern "C" void kernel_launch(void* const* d_in, const int* in_sizes, int n_in,
                              void* d_out, int out_size, void* d_ws, size_t ws_size,
                              hipStream_t stream) {
  (void)in_sizes; (void)n_in; (void)out_size; (void)ws_size;
  const float* inputs = (const float*)d_in[0];   // (B,H)
  const float* h0     = (const float*)d_in[1];   // (B,H)
  const float* W_ih   = (const float*)d_in[2];   // (3H, H+1)
  const float* W_hh   = (const float*)d_in[3];   // (3H, H)
  const float* b_ih   = (const float*)d_in[4];
  const float* b_hh   = (const float*)d_in[5];
  const float* w_halt = (const float*)d_in[6];
  const float* b_halt = (const float*)d_in[7];

  float* out       = (float*)d_out;
  float* merged    = out;                               // (B,H)
  float* mergedOut = out + (size_t)BDIM * HDIM;         // (B,H) copy
  float* hpAcc     = out + 2 * (size_t)BDIM * HDIM;     // (B,)
  float* numUpd    = hpAcc + BDIM;                      // (B,)

  char* w = (char*)d_ws;
  ushort_t* inBf  = (ushort_t*)w;  w += (size_t)BDIM * HDIM * 2;
  ushort_t* WihBf = (ushort_t*)w;  w += (size_t)GDIM * HDIM * 2;
  ushort_t* WhhBf = (ushort_t*)w;  w += (size_t)GDIM * HDIM * 2;
  float*    Gi    = (float*)w;     w += (size_t)BDIM * GDIM * 4;
  float*    hF[2]; hF[0] = (float*)w; w += (size_t)BDIM * HDIM * 4;
                   hF[1] = (float*)w; w += (size_t)BDIM * HDIM * 4;
  ushort_t* hB[2]; hB[0] = (ushort_t*)w; w += (size_t)BDIM * HDIM * 2;
                   hB[1] = (ushort_t*)w; w += (size_t)BDIM * HDIM * 2;
  float*    accum = (float*)w;     w += (size_t)BDIM * 4;

  const int nBH = BDIM * HDIM;        // 4,194,304
  const int nGH = GDIM * HDIM;        // 3,145,728
  const int gBH = (nBH + 255) / 256;
  const int gGH = (nGH + 255) / 256;
  const int gB  = (BDIM + 255) / 256;

  // deterministic init (d_out/d_ws are poisoned by the harness)
  k_zero<<<gBH, 256, 0, stream>>>(merged, nBH);
  k_zero<<<gB, 256, 0, stream>>>(hpAcc, BDIM);
  k_zero<<<gB, 256, 0, stream>>>(numUpd, BDIM);
  k_zero<<<gB, 256, 0, stream>>>(accum, BDIM);

  k_cvt<<<gBH, 256, 0, stream>>>(inputs, inBf, nBH);
  k_cvt<<<gGH, 256, 0, stream>>>(W_hh, WhhBf, nGH);
  k_cvt_wih<<<gGH, 256, 0, stream>>>(W_ih, WihBf);
  k_inith<<<gBH, 256, 0, stream>>>(h0, hF[0], hB[0], nBH);

  const dim3 gGemm(HDIM / 32, BDIM / 256);   // (32, 16)

  // Gi_base = inputs @ W_ih[:, :H]^T + b_ih (flag column folded in per-step)
  k_gemm<<<gGemm, 256, 0, stream>>>(inBf, WihBf, 0, 0.f,
                                    Gi, b_ih,
                                    nullptr, nullptr, nullptr, nullptr,
                                    nullptr, nullptr);

  int cur = 0;
  for (int d = 0; d < DEPTH; ++d) {
    k_halt<<<BDIM / 8, 256, 0, stream>>>(hF[cur], w_halt, b_halt, accum,
                                         merged, hpAcc, numUpd, d);
    const float flag = (d <= 1) ? 1.f : 0.f;
    const int nxt = cur ^ 1;
    k_gemm<<<gGemm, 256, 0, stream>>>(hB[cur], WhhBf, 1, flag,
                                      nullptr, nullptr,
                                      Gi, b_hh, W_ih, hF[cur],
                                      hF[nxt], hB[nxt]);
    cur = nxt;
  }

  k_copy<<<gBH, 256, 0, stream>>>(mergedOut, merged, nBH);   // merged_output == merged_hidden
}
